// DynamicComposeBlock_65798898974902
// MI455X (gfx1250) — matrix operand order repaired and checked by reading, not compiled, not run on hardware
//
#include <hip/hip_runtime.h>
#include <hip/hip_bf16.h>

// Problem constants (from reference setup_inputs)
#define BB 2
#define CC 256
#define OO 256
#define TT 16
#define HH 64
#define WW 64
#define HWD (HH * WW)          // 4096

typedef __attribute__((ext_vector_type(16))) __bf16 v16bf;
typedef __attribute__((ext_vector_type(8)))  __bf16 v8bf;
typedef __attribute__((ext_vector_type(4)))  __bf16 v4bf;
typedef __attribute__((ext_vector_type(8)))  float  v8f;
typedef __attribute__((ext_vector_type(4)))  float  v4f;

// GCC-style vector to match the builtin's parameter type exactly:
// 'int __attribute__((vector_size(16)))'
typedef int v4i_vs __attribute__((vector_size(16)));

#define GLOBAL_AS __attribute__((address_space(1)))
#define LDS_AS    __attribute__((address_space(3)))

#if defined(__gfx1250__) && __has_builtin(__builtin_amdgcn_global_load_async_to_lds_b128)
#define HAVE_ASYNC_LDS 1
#else
#define HAVE_ASYNC_LDS 0
#endif

// 16-byte global -> LDS copy; async (ASYNCcnt-tracked) when available.
__device__ __forceinline__ void copy16_g2l(const __bf16* __restrict__ src,
                                           __bf16* __restrict__ dst_lds) {
#if HAVE_ASYNC_LDS
    const GLOBAL_AS void* gv = (const GLOBAL_AS void*)src;   // known-good cast
    LDS_AS void*          lv = (LDS_AS void*)dst_lds;
    __builtin_amdgcn_global_load_async_to_lds_b128(
        (GLOBAL_AS v4i_vs*)gv, (LDS_AS v4i_vs*)lv, 0, 0);
#else
    *reinterpret_cast<v8bf*>(dst_lds) = *reinterpret_cast<const v8bf*>(src);
#endif
}

__device__ __forceinline__ void wait_async_copies() {
#if HAVE_ASYNC_LDS
#if __has_builtin(__builtin_amdgcn_s_wait_asynccnt)
    __builtin_amdgcn_s_wait_asynccnt(0);
#else
    asm volatile("s_wait_asynccnt 0x0" ::: "memory");
#endif
#endif
}

// ---------------- Pre-pass kernels (run once per launch) ----------------

// W3d fp32 [O,C] -> bf16 [O,C].  grid 64 x 256 threads, 4 elems/thread.
__global__ __launch_bounds__(256) void pre_convert_w3d(
    const float* __restrict__ W3d, __bf16* __restrict__ w3d_bf) {
    int i4 = blockIdx.x * 256 + threadIdx.x;        // float4 index, 0..16383
    v4f v = reinterpret_cast<const v4f*>(W3d)[i4];
    v4bf p;
    p[0] = (__bf16)v[0]; p[1] = (__bf16)v[1];
    p[2] = (__bf16)v[2]; p[3] = (__bf16)v[3];
    reinterpret_cast<v4bf*>(w3d_bf)[i4] = p;
}

// fea_th [B,C,T,H] -> th_bf [B,T,H,C] bf16; fea_tw likewise with W.
// grid 4096 x 256: first 2048 blocks handle th, rest handle tw.
__global__ __launch_bounds__(256) void pre_transpose(
    const float* __restrict__ fea_th, const float* __restrict__ fea_tw,
    __bf16* __restrict__ th_bf, __bf16* __restrict__ tw_bf) {
    bool is_tw = blockIdx.x >= 2048;
    int gid = (is_tw ? blockIdx.x - 2048 : blockIdx.x) * 256 + threadIdx.x; // 0..524287
    int c = gid & 255;
    int x = (gid >> 8) & 63;        // h or w
    int t = (gid >> 14) & 15;
    int b = gid >> 18;
    const float* src = is_tw ? fea_tw : fea_th;
    __bf16* dst = is_tw ? tw_bf : th_bf;
    dst[gid] = (__bf16)src[((b * CC + c) * TT + t) * 64 + x];
}

// obj[b,t,o] = b1d[o] + sum_c W1d[o,c] * fea_obj[b,c,t].  grid 32 x 256.
__global__ __launch_bounds__(256) void pre_obj(
    const float* __restrict__ fea_obj, const float* __restrict__ W1d,
    const float* __restrict__ b1d, float* __restrict__ obj_ws) {
    __shared__ float fo[CC];
    const int tid = threadIdx.x;
    const int bt = blockIdx.x;                 // b*16 + t
    const int b = bt >> 4, t = bt & 15;
    fo[tid] = fea_obj[(b * CC + tid) * TT + t];
    __syncthreads();
    float a = b1d[tid];
    const v4f* wr = reinterpret_cast<const v4f*>(W1d + tid * CC);
    #pragma unroll 8
    for (int i = 0; i < 64; ++i) {
        v4f wv = wr[i];
        const float* f = &fo[i * 4];
        a += wv[0] * f[0] + wv[1] * f[1] + wv[2] * f[2] + wv[3] * f[3];
    }
    obj_ws[bt * OO + tid] = a;
}

// ---------------- Main WMMA kernel ----------------
// One workgroup = 256 threads = 8 wave32; tile = all 256 o x (8h x 16w) for one (b,t).
__global__ __launch_bounds__(256) void dyn_compose_wmma_kernel(
    const __bf16* __restrict__ w3d_bf,  // [O,C] bf16
    const __bf16* __restrict__ th_bf,   // [B,T,H,C] bf16
    const __bf16* __restrict__ tw_bf,   // [B,T,W,C] bf16
    const float*  __restrict__ obj_g,   // [B,T,O] fp32
    const float*  __restrict__ heatmap, // [B,1,T,H,W]
    const float*  __restrict__ mask,    // [B,1,T,H,W]
    const float*  __restrict__ b3d,     // [O]
    float* __restrict__ out)            // [B,O,T,H,W]
{
    __shared__ __align__(16) __bf16 w3d_lds[OO * CC];   // 131072 B, [o][c]
    __shared__ __align__(16) __bf16 tw_lds[16 * CC];    //   8192 B, [wsub][c]
    __shared__ __align__(16) __bf16 th_lds[8 * CC];     //   4096 B, [hsub][c]
    __shared__ __align__(16) float  obj_lds[OO];        //   1024 B
    __shared__ __align__(16) float  b3d_lds[OO];        //   1024 B

    const int tid = threadIdx.x;
    const int bx  = blockIdx.x;           // 1024 blocks: b(2)*t(16)*hblk(8)*wblk(4)
    const int w0  = (bx & 3) * 16;
    const int h0  = ((bx >> 2) & 7) * 8;
    const int t   = (bx >> 5) & 15;
    const int b   = bx >> 9;

    // ---- Phase 0: pure byte staging (async global->LDS when available) ----
    // W3d bf16: 8192 granules of 16B -> 32 per thread
    #pragma unroll 4
    for (int i = 0; i < 32; ++i) {
        int e = (i * 256 + tid) * 8;
        copy16_g2l(w3d_bf + e, &w3d_lds[e]);
    }
    // tw rows [w0, w0+16): 4096 contiguous bf16 (both sides c-major)
    {
        const __bf16* src = tw_bf + ((b * TT + t) * WW + w0) * CC;
        #pragma unroll
        for (int i = 0; i < 2; ++i) {
            int e = (i * 256 + tid) * 8;
            copy16_g2l(src + e, &tw_lds[e]);
        }
    }
    // th rows [h0, h0+8): 2048 contiguous bf16
    {
        const __bf16* src = th_bf + ((b * TT + t) * HH + h0) * CC;
        int e = tid * 8;
        copy16_g2l(src + e, &th_lds[e]);
    }
    // small fp32 tables
    obj_lds[tid] = obj_g[(b * TT + t) * OO + tid];
    b3d_lds[tid] = b3d[tid];
    wait_async_copies();
    __syncthreads();

    // ---- Phase 1: WMMA main loop ----
    const int wv   = tid >> 5;        // wave id 0..7 -> h-row
    const int lane = tid & 31;
    const int lo   = lane & 15;       // column within tile / M-row select
    const int hi   = lane >> 4;       // half-wave select

    v8f acc[16];
    #pragma unroll
    for (int ot = 0; ot < 16; ++ot) acc[ot] = (v8f){0,0,0,0,0,0,0,0};

    const __bf16* throw_p = &th_lds[wv * CC];
    const __bf16* twcol_p = &tw_lds[lo * CC];

    for (int kk = 0; kk < 8; ++kk) {
        const int kb = kk * 32;

        // B fragment (32x16 bf16): element e <-> K = hi*16 + e, column N = lo
        const int c0 = kb + hi * 16;
        v8bf th0 = *reinterpret_cast<const v8bf*>(&throw_p[c0]);
        v8bf th1 = *reinterpret_cast<const v8bf*>(&throw_p[c0 + 8]);
        v8bf tw0 = *reinterpret_cast<const v8bf*>(&twcol_p[c0]);
        v8bf tw1 = *reinterpret_cast<const v8bf*>(&twcol_p[c0 + 8]);
        v16bf bfrag;
        #pragma unroll
        for (int e = 0; e < 8; ++e) {
            bfrag[e]     = (__bf16)((float)th0[e] * (float)tw0[e]);
            bfrag[e + 8] = (__bf16)((float)th1[e] * (float)tw1[e]);
        }

        // 16 o-tiles: A fragment (16x32 bf16): elems 0-7 <-> K = hi*16 + e,
        // elems 8-15 <-> K = hi*16 + 8 + e; M-row = lo.
        #pragma unroll
        for (int ot = 0; ot < 16; ++ot) {
            const int o    = ot * 16 + lo;
            const int base = o * CC + kb + hi * 16;
            v8bf a0 = *reinterpret_cast<const v8bf*>(&w3d_lds[base]);
            v8bf a1 = *reinterpret_cast<const v8bf*>(&w3d_lds[base + 8]);
            v16bf afrag;
            #pragma unroll
            for (int e = 0; e < 8; ++e) { afrag[e] = a0[e]; afrag[e + 8] = a1[e]; }
            acc[ot] = __builtin_amdgcn_wmma_f32_16x16x32_bf16(
                false, afrag, false, bfrag, (short)0, acc[ot], false, false);
        }
    }

    // ---- Phase 2: fused epilogue ----
    // D layout: VGPR r -> M = r + hi*8, N = lo.
    const int hrow = h0 + wv;
    const int wcol = w0 + lo;
    const int hw   = hrow * WW + wcol;
    const int hti  = (b * TT + t) * HWD + hw;   // [B,1,T,H,W] flat
    const float hm = heatmap[hti];
    const float mk = mask[hti];
    const float sA = (1.0f - hm) * mk;
    const float sB = hm * mk;

    #pragma unroll
    for (int ot = 0; ot < 16; ++ot) {
        #pragma unroll
        for (int r = 0; r < 8; ++r) {
            const int o = ot * 16 + hi * 8 + r;
            const float val = (acc[ot][r] + b3d_lds[o]) * sA + obj_lds[o] * sB;
            out[((b * OO + o) * TT + t) * HWD + hw] = val;
        }
    }
}

extern "C" void kernel_launch(void* const* d_in, const int* in_sizes, int n_in,
                              void* d_out, int out_size, void* d_ws, size_t ws_size,
                              hipStream_t stream) {
    (void)in_sizes; (void)n_in; (void)out_size; (void)ws_size;
    const float* fea_th  = (const float*)d_in[0];
    const float* fea_tw  = (const float*)d_in[1];
    const float* fea_obj = (const float*)d_in[2];
    const float* heatmap = (const float*)d_in[3];
    const float* mask    = (const float*)d_in[4];
    const float* W3d     = (const float*)d_in[5];
    const float* b3d     = (const float*)d_in[6];
    const float* W1d     = (const float*)d_in[7];
    const float* b1d     = (const float*)d_in[8];
    float* out = (float*)d_out;

    // Workspace layout (bytes): w3d_bf 131072 | th_bf 1048576 | tw_bf 1048576 | obj 32768
    __bf16* w3d_bf = (__bf16*)d_ws;
    __bf16* th_bf  = w3d_bf + OO * CC;                 // +65536 elems
    __bf16* tw_bf  = th_bf + BB * TT * HH * CC;        // +524288 elems
    float*  obj_ws = (float*)(tw_bf + BB * TT * WW * CC);

    pre_convert_w3d<<<dim3(64),   dim3(256), 0, stream>>>(W3d, w3d_bf);
    pre_transpose  <<<dim3(4096), dim3(256), 0, stream>>>(fea_th, fea_tw, th_bf, tw_bf);
    pre_obj        <<<dim3(32),   dim3(256), 0, stream>>>(fea_obj, W1d, b1d, obj_ws);

    dyn_compose_wmma_kernel<<<dim3(BB * TT * 8 * 4), dim3(256), 0, stream>>>(
        w3d_bf, th_bf, tw_bf, obj_ws, heatmap, mask, b3d, out);
}
